// HRWKV_7_50972671869168
// MI455X (gfx1250) — matrix operand order repaired and checked by reading, not compiled, not run on hardware
//
#include <hip/hip_runtime.h>
#include <hip/hip_bf16.h>

// ---------------------------------------------------------------------------
// Shapes (compile-time constants from the reference)
// ---------------------------------------------------------------------------
#define BB   2
#define TT   1024
#define HH   32
#define NN   64
#define HNN  2048          // H*N
#define KVH_ 8
#define KVD_ 512
#define MM   (BB*TT)       // 2048 token rows
#define RKVN 3072          // HN + 2*KVD
#define SCHUNK 8           // timesteps staged per barrier in the scan

typedef __attribute__((ext_vector_type(16))) __bf16 v16bf;
typedef __attribute__((ext_vector_type(8)))  __bf16 v8bf;
typedef __attribute__((ext_vector_type(8)))  float  v8f;

// ---------------------------------------------------------------------------
// small device helpers
// ---------------------------------------------------------------------------
__device__ __forceinline__ float wred32(float v) {
  #pragma unroll
  for (int m = 16; m > 0; m >>= 1) v += __shfl_xor(v, m, 32);
  return v;
}
__device__ __forceinline__ float sigm(float x) { return 1.0f / (1.0f + __expf(-x)); }
__device__ __forceinline__ float softplus_f(float x) {
  return fmaxf(x, 0.0f) + log1pf(__expf(-fabsf(x)));
}

// ---------------------------------------------------------------------------
// weight / activation casts
// ---------------------------------------------------------------------------
__global__ void cast_bf16_kernel(const float* __restrict__ s, __bf16* __restrict__ d, int n) {
  int i = blockIdx.x * 256 + threadIdx.x;
  if (i < n) d[i] = (__bf16)s[i];
}
// src is (K rows, N cols) row-major; dst becomes (N rows, K cols) => NT form
__global__ void cast_bf16_tr_kernel(const float* __restrict__ s, __bf16* __restrict__ d,
                                    int K, int N) {
  int i = blockIdx.x * 256 + threadIdx.x;
  if (i < K * N) {
    int k = i / N, n = i % N;
    d[(size_t)n * K + k] = (__bf16)s[i];
  }
}
__global__ void concat_bias_kernel(const float* __restrict__ a, const float* __restrict__ b,
                                   const float* __restrict__ c, float* __restrict__ o) {
  int i = blockIdx.x * 256 + threadIdx.x;
  if (i < HNN)                o[i] = a[i];
  else if (i < HNN + KVD_)    o[i] = b[i - HNN];
  else if (i < RKVN)          o[i] = c[i - HNN - KVD_];
}

// ---------------------------------------------------------------------------
// RMS norm of x_in over HN, emit bf16 activations
// ---------------------------------------------------------------------------
__global__ __launch_bounds__(256) void rms_x_kernel(const float* __restrict__ xin,
                                                    const float* __restrict__ ln1,
                                                    __bf16* __restrict__ xbf) {
  __shared__ float red[256];
  const int m = blockIdx.x;
  const float* row = xin + (size_t)m * HNN;
  float s = 0.0f;
  for (int i = threadIdx.x; i < HNN; i += 256) { float v = row[i]; s += v * v; }
  red[threadIdx.x] = s;
  __syncthreads();
  for (int st = 128; st > 0; st >>= 1) {
    if (threadIdx.x < st) red[threadIdx.x] += red[threadIdx.x + st];
    __syncthreads();
  }
  float inv = rsqrtf(red[0] * (1.0f / (float)HNN) + 1e-6f);
  for (int i = threadIdx.x; i < HNN; i += 256)
    xbf[(size_t)m * HNN + i] = (__bf16)(ln1[i] * row[i] * inv);
}

// ---------------------------------------------------------------------------
// WMMA bf16 GEMM (NT):  C[M,N] = A[M,K] * W[N,K]^T (+bias[n]) (+addsrc[m,n])
// register-blocked: one wave -> 32x32 output (2x2 fragments, 4 WMMA / K-chunk)
// block = 4 waves (2x2) -> 64x64 tile; requires M%64==0, N%64==0, K%32==0
// ---------------------------------------------------------------------------
__global__ __launch_bounds__(128) void gemm_bf16_nt(const __bf16* __restrict__ A, int lda,
                                                    const __bf16* __restrict__ W, int ldw,
                                                    const float* __restrict__ bias,
                                                    const float* __restrict__ addsrc,
                                                    float* __restrict__ C, int ldc, int K) {
  const int lane = threadIdx.x & 31;
  const int wave = threadIdx.x >> 5;   // 0..3
  const int wm   = wave >> 1;          // 0..1
  const int wn   = wave & 1;           // 0..1
  const int tm   = blockIdx.y * 64 + wm * 32;
  const int tn   = blockIdx.x * 64 + wn * 32;
  const int hsel = lane >> 4;          // 0: lanes 0-15, 1: lanes 16-31
  const int l15  = lane & 15;

  const __bf16* arow0 = A + (size_t)(tm + l15) * lda;
  const __bf16* arow1 = A + (size_t)(tm + 16 + l15) * lda;
  const __bf16* wrow0 = W + (size_t)(tn + l15) * ldw;
  const __bf16* wrow1 = W + (size_t)(tn + 16 + l15) * ldw;

  v8f acc00 = {}, acc01 = {}, acc10 = {}, acc11 = {};
  for (int k0 = 0; k0 < K; k0 += 32) {
    if (k0 + 128 < K) {                       // uniform branch: prefetch ahead
      __builtin_prefetch(arow0 + k0 + 128, 0, 3);
      __builtin_prefetch(arow1 + k0 + 128, 0, 3);
      __builtin_prefetch(wrow0 + k0 + 128, 0, 3);
      __builtin_prefetch(wrow1 + k0 + 128, 0, 3);
    }
    // A 16x32 bf16 fragment: lanes 0-15 hold K 0..15, lanes 16-31 hold K 16..31
    union { v16bf v; v8bf h[2]; } a0, a1;
    a0.h[0] = *(const v8bf*)(arow0 + k0 + hsel * 16);
    a0.h[1] = *(const v8bf*)(arow0 + k0 + hsel * 16 + 8);
    a1.h[0] = *(const v8bf*)(arow1 + k0 + hsel * 16);
    a1.h[1] = *(const v8bf*)(arow1 + k0 + hsel * 16 + 8);
    // B 32x16 bf16 fragment: lanes 0-15 hold K 0..15, lanes 16-31 hold K 16..31
    v16bf b0 = *(const v16bf*)(wrow0 + k0 + hsel * 16);
    v16bf b1 = *(const v16bf*)(wrow1 + k0 + hsel * 16);
    acc00 = __builtin_amdgcn_wmma_f32_16x16x32_bf16(false, a0.v, false, b0, (short)0, acc00, false, false);
    acc01 = __builtin_amdgcn_wmma_f32_16x16x32_bf16(false, a0.v, false, b1, (short)0, acc01, false, false);
    acc10 = __builtin_amdgcn_wmma_f32_16x16x32_bf16(false, a1.v, false, b0, (short)0, acc10, false, false);
    acc11 = __builtin_amdgcn_wmma_f32_16x16x32_bf16(false, a1.v, false, b1, (short)0, acc11, false, false);
  }

  auto store_tile = [&](const v8f& acc, int rbase, int cbase) {
    const float bv = bias ? bias[cbase + l15] : 0.0f;
    #pragma unroll
    for (int i = 0; i < 8; ++i) {
      const int row = rbase + i + hsel * 8;  // C/D layout: VGPR i -> M=i (lo) / M=8+i (hi)
      const int col = cbase + l15;
      float v = acc[i] + bv;
      if (addsrc) v += addsrc[(size_t)row * ldc + col];
      C[(size_t)row * ldc + col] = v;
    }
  };
  store_tile(acc00, tm,      tn);
  store_tile(acc01, tm,      tn + 16);
  store_tile(acc10, tm + 16, tn);
  store_tile(acc11, tm + 16, tn + 16);
}

// ---------------------------------------------------------------------------
// split x@wavgk1 into its five LoRA inputs with pre-GEMM activations
// ---------------------------------------------------------------------------
__global__ void act_split_kernel(const float* __restrict__ xwavgk,
                                 __bf16* __restrict__ xw_bf, __bf16* __restrict__ xa_bf,
                                 __bf16* __restrict__ xv_bf, __bf16* __restrict__ xg_bf,
                                 __bf16* __restrict__ xk_bf) {
  int idx = blockIdx.x * 256 + threadIdx.x;
  if (idx >= MM * 320) return;
  int m = idx / 320, c = idx % 320;
  float v = xwavgk[idx];
  if      (c < 64)  xw_bf[(size_t)m * 64  + c]        = (__bf16)tanhf(v);
  else if (c < 128) xa_bf[(size_t)m * 64  + (c - 64)] = (__bf16)v;
  else if (c < 160) xv_bf[(size_t)m * 32  + (c - 128)] = (__bf16)v;
  else if (c < 288) xg_bf[(size_t)m * 128 + (c - 160)] = (__bf16)sigm(v);
  else              xk_bf[(size_t)m * 32  + (c - 288)] = (__bf16)v;
}

// ---------------------------------------------------------------------------
// per-token prep: RMS+RoPE on r/k, first-token gating, w/a, kk, aa/bb, bonus
// one block per token; wave w handles heads 4w..4w+3; lane holds n and n+32
// ---------------------------------------------------------------------------
__global__ __launch_bounds__(256) void prep_kernel(
    const float* __restrict__ rkv,    const float* __restrict__ w_pre,
    const float* __restrict__ a_pre,  const float* __restrict__ gv_pre,
    const float* __restrict__ gk_pre, const float* __restrict__ v_first,
    const float* __restrict__ k_first,const float* __restrict__ ccos,
    const float* __restrict__ csin,
    const float* __restrict__ Rb, const float* __restrict__ Kb, const float* __restrict__ Vb,
    const float* __restrict__ ln_r, const float* __restrict__ ln_k,
    const float* __restrict__ r_k,
    float* __restrict__ r_o, float* __restrict__ decay_o, float* __restrict__ k_o,
    float* __restrict__ v_o, float* __restrict__ aa_o, float* __restrict__ bb_o,
    float* __restrict__ bonus_o) {
  const int m = blockIdx.x;
  const int wave = threadIdx.x >> 5, lane = threadIdx.x & 31;
  const int n0 = lane, n1 = lane + 32;
  const float c0 = ccos[(size_t)m * NN + n0], c1 = ccos[(size_t)m * NN + n1];
  const float s0 = csin[(size_t)m * NN + n0], s1 = csin[(size_t)m * NN + n1];

  for (int hh = 0; hh < 4; ++hh) {
    const int h = wave * 4 + hh;
    const int kvh = h >> 2;
    const size_t rbase = (size_t)m * RKVN + (size_t)h * NN;
    const size_t kbase = (size_t)m * RKVN + HNN + (size_t)kvh * NN;
    const size_t vbase = (size_t)m * RKVN + HNN + KVD_ + (size_t)kvh * NN;
    const size_t kvoff = (size_t)m * KVD_ + (size_t)kvh * NN;
    const size_t hn0 = (size_t)m * HNN + (size_t)h * NN + n0;
    const size_t hn1 = hn0 + 32;

    // r: bias + RMS(ln_r) + RoPE
    float r0 = rkv[rbase + n0] + Rb[h * NN + n0];
    float r1 = rkv[rbase + n1] + Rb[h * NN + n1];
    float rinv = rsqrtf(wred32(r0 * r0 + r1 * r1) * (1.0f / NN) + 1e-6f);
    r0 = ln_r[n0] * r0 * rinv;  r1 = ln_r[n1] * r1 * rinv;
    const float rr0 = r0 * c0 - r1 * s0;
    const float rr1 = r1 * c1 + r0 * s1;

    // k: bias + RMS(ln_k) + RoPE + first-token gate
    float kv0 = rkv[kbase + n0] + Kb[kvh * NN + n0];
    float kv1 = rkv[kbase + n1] + Kb[kvh * NN + n1];
    float kinv = rsqrtf(wred32(kv0 * kv0 + kv1 * kv1) * (1.0f / NN) + 1e-6f);
    kv0 = ln_k[n0] * kv0 * kinv;  kv1 = ln_k[n1] * kv1 * kinv;
    float kk0 = kv0 * c0 - kv1 * s0;
    float kk1 = kv1 * c1 + kv0 * s1;
    const float gk0 = sigm(gk_pre[kvoff + n0]), gk1 = sigm(gk_pre[kvoff + n1]);
    kk0 += (k_first[kvoff + n0] - kk0) * gk0;
    kk1 += (k_first[kvoff + n1] - kk1) * gk1;

    // v: bias + first-token gate
    float vv0 = rkv[vbase + n0] + Vb[kvh * NN + n0];
    float vv1 = rkv[vbase + n1] + Vb[kvh * NN + n1];
    const float gv0 = sigm(gv_pre[kvoff + n0]), gv1 = sigm(gv_pre[kvoff + n1]);
    vv0 += (v_first[kvoff + n0] - vv0) * gv0;
    vv1 += (v_first[kvoff + n1] - vv1) * gv1;

    // w (log-decay) and a (in-context lr)
    const float w0v = -softplus_f(-w_pre[hn0]) - 0.5f;
    const float w1v = -softplus_f(-w_pre[hn1]) - 0.5f;
    const float a0v = sigm(a_pre[hn0]);
    const float a1v = sigm(a_pre[hn1]);

    // kk normalization (of gated k), removal/replacement vectors
    const float nrm = sqrtf(wred32(kk0 * kk0 + kk1 * kk1));
    const float ninv = 1.0f / fmaxf(nrm, 1e-12f);
    const float u0 = kk0 * ninv, u1 = kk1 * ninv;
    const float kf0 = kk0 * (1.0f - w0v + a0v);
    const float kf1 = kk1 * (1.0f - w1v + a1v);

    // bonus = (sum_n r*k*r_k) * v
    const float dot = wred32(rr0 * kf0 * r_k[h * NN + n0] + rr1 * kf1 * r_k[h * NN + n1]);

    r_o[hn0] = rr0;  r_o[hn1] = rr1;
    decay_o[hn0] = __expf(-__expf(w0v));  decay_o[hn1] = __expf(-__expf(w1v));
    k_o[hn0] = kf0;  k_o[hn1] = kf1;
    v_o[hn0] = vv0;  v_o[hn1] = vv1;
    aa_o[hn0] = -u0; aa_o[hn1] = -u1;
    bb_o[hn0] = u0 * a0v; bb_o[hn1] = u1 * a1v;
    bonus_o[hn0] = dot * vv0; bonus_o[hn1] = dot * vv1;
  }
}

// ---------------------------------------------------------------------------
// sequential state scan: one block per (b,h); 64x64 fp32 state in registers.
// SCHUNK timesteps of per-step vectors are staged into LDS per barrier via the
// CDNA5 async memory->LDS path (ASYNCcnt), then consumed barrier-free.
// thread owns row v = tid/4, cols [(tid&3)*16 .. +15]; 4-lane shfl reductions
// ---------------------------------------------------------------------------
__global__ __launch_bounds__(256) void scan_kernel(
    const float* __restrict__ rbuf, const float* __restrict__ decay,
    const float* __restrict__ kbuf, const float* __restrict__ vbuf,
    const float* __restrict__ aabuf, const float* __restrict__ bbbuf,
    const float* __restrict__ state0, float* __restrict__ obuf) {
  const int bh = blockIdx.x;            // b*H + h
  const int b = bh >> 5, h = bh & 31;
  const int tid = threadIdx.x;
  const int vrow = tid >> 2;
  const int kgrp = tid & 3;
  const int kbase = kgrp * 16;

  float S[16];
  const float* st = state0 + (size_t)bh * (NN * NN);
  #pragma unroll
  for (int j = 0; j < 16; ++j) S[j] = st[vrow * NN + kbase + j];

  __shared__ float ls[SCHUNK * 6 * NN];      // per step: r, decay, k, aa, bb, v
  const size_t base = ((size_t)b * TT) * HNN + (size_t)h * NN;

  for (int tc = 0; tc < TT; tc += SCHUNK) {
    const size_t coff = base + (size_t)tc * HNN;
    // stage SCHUNK*6*64 floats: async global->LDS, no VGPR bounce
    for (int i = tid; i < SCHUNK * 6 * NN; i += 256) {
      const int s   = i / (6 * NN);
      const int rem = i % (6 * NN);
      const int sel = rem >> 6, n = rem & 63;
      const float* p = (sel == 0) ? rbuf : (sel == 1) ? decay : (sel == 2) ? kbuf
                      : (sel == 3) ? aabuf : (sel == 4) ? bbbuf : vbuf;
      const float* gp = p + coff + (size_t)s * HNN + n;
      // generic pointer to __shared__: low 32 bits are the LDS byte address
      unsigned la = (unsigned)(uintptr_t)(&ls[i]);
      asm volatile("global_load_async_to_lds_b32 %0, %1, off"
                   :: "v"(la), "v"(gp) : "memory");
    }
    asm volatile("s_wait_asynccnt 0x0" ::: "memory");
    __syncthreads();

    for (int s = 0; s < SCHUNK; ++s) {
      const float* L = &ls[s * 6 * NN];
      float sa = 0.0f;
      #pragma unroll
      for (int j = 0; j < 16; ++j) sa += S[j] * L[3 * NN + kbase + j];    // aa
      sa += __shfl_xor(sa, 1, 32);
      sa += __shfl_xor(sa, 2, 32);

      const float vt = L[5 * NN + vrow];
      float o = 0.0f;
      #pragma unroll
      for (int j = 0; j < 16; ++j) {
        const int kc = kbase + j;
        const float sv = S[j] * L[NN + kc] + sa * L[4 * NN + kc] + vt * L[2 * NN + kc];
        S[j] = sv;
        o += sv * L[kc];                                                  // r
      }
      o += __shfl_xor(o, 1, 32);
      o += __shfl_xor(o, 2, 32);
      if (kgrp == 0) obuf[coff + (size_t)s * HNN + vrow] = o;
    }
    __syncthreads();
  }
}

// ---------------------------------------------------------------------------
// y = bf16((o * N^-0.5 + bonus) * g)   (input to the final O_w GEMM)
// ---------------------------------------------------------------------------
__global__ void mixg_kernel(const float* __restrict__ o, const float* __restrict__ bonus,
                            const float* __restrict__ g, __bf16* __restrict__ y, int n) {
  int i = blockIdx.x * 256 + threadIdx.x;
  if (i < n) y[i] = (__bf16)((o[i] * 0.125f + bonus[i]) * g[i]);
}

// ---------------------------------------------------------------------------
// host launcher
// ---------------------------------------------------------------------------
static inline void launch_gemm(const __bf16* A, int lda, const __bf16* W, int ldw,
                               const float* bias, const float* addsrc,
                               float* C, int ldc, int K, int N, hipStream_t s) {
  dim3 grid(N / 64, MM / 64);
  gemm_bf16_nt<<<grid, 128, 0, s>>>(A, lda, W, ldw, bias, addsrc, C, ldc, K);
}

extern "C" void kernel_launch(void* const* d_in, const int* in_sizes, int n_in,
                              void* d_out, int out_size, void* d_ws, size_t ws_size,
                              hipStream_t stream) {
  const float* x_in    = (const float*)d_in[0];
  const float* v_first = (const float*)d_in[1];
  const float* k_first = (const float*)d_in[2];
  const float* state   = (const float*)d_in[3];
  const float* ccos    = (const float*)d_in[4];
  const float* csin    = (const float*)d_in[5];
  const float* wavgk1  = (const float*)d_in[6];
  const float* w0      = (const float*)d_in[7];
  const float* w2      = (const float*)d_in[8];
  const float* a0      = (const float*)d_in[9];
  const float* a2      = (const float*)d_in[10];
  const float* v0      = (const float*)d_in[11];
  const float* v2      = (const float*)d_in[12];
  const float* g2      = (const float*)d_in[13];
  const float* k0      = (const float*)d_in[14];
  const float* k2      = (const float*)d_in[15];
  const float* r_k     = (const float*)d_in[16];
  const float* RKV_w   = (const float*)d_in[17];
  const float* O_w     = (const float*)d_in[18];
  const float* R_bias  = (const float*)d_in[19];
  const float* K_bias  = (const float*)d_in[20];
  const float* V_bias  = (const float*)d_in[21];
  const float* O_bias  = (const float*)d_in[22];
  const float* ln_r    = (const float*)d_in[23];
  const float* ln_k    = (const float*)d_in[24];
  const float* ln1     = (const float*)d_in[25];
  float* out = (float*)d_out;

  char* ws = (char*)d_ws;
  size_t off = 0;
  auto alloc = [&](size_t bytes) -> void* {
    void* p = ws + off;
    off = (off + bytes + 255) & ~(size_t)255;
    return p;
  };

  // bf16 buffers
  __bf16* xbf     = (__bf16*)alloc((size_t)MM * HNN * 2);
  __bf16* wrkv_bf = (__bf16*)alloc((size_t)RKVN * HNN * 2);
  __bf16* wavgT   = (__bf16*)alloc((size_t)320 * HNN * 2);
  __bf16* w2T     = (__bf16*)alloc((size_t)HNN * 64 * 2);
  __bf16* a2T     = (__bf16*)alloc((size_t)HNN * 64 * 2);
  __bf16* v2T     = (__bf16*)alloc((size_t)KVD_ * 32 * 2);
  __bf16* g2T     = (__bf16*)alloc((size_t)HNN * 128 * 2);
  __bf16* k2T     = (__bf16*)alloc((size_t)KVD_ * 32 * 2);
  __bf16* owbf    = (__bf16*)alloc((size_t)HNN * HNN * 2);
  __bf16* xw_bf   = (__bf16*)alloc((size_t)MM * 64 * 2);
  __bf16* xa_bf   = (__bf16*)alloc((size_t)MM * 64 * 2);
  __bf16* xv_bf   = (__bf16*)alloc((size_t)MM * 32 * 2);
  __bf16* xg_bf   = (__bf16*)alloc((size_t)MM * 128 * 2);
  __bf16* xk_bf   = (__bf16*)alloc((size_t)MM * 32 * 2);
  __bf16* ybf     = (__bf16*)alloc((size_t)MM * HNN * 2);
  // fp32 buffers
  float* biascat = (float*)alloc((size_t)RKVN * 4);
  float* rkv     = (float*)alloc((size_t)MM * RKVN * 4);
  float* xwavgk  = (float*)alloc((size_t)MM * 320 * 4);
  float* w_pre   = (float*)alloc((size_t)MM * HNN * 4);
  float* a_pre   = (float*)alloc((size_t)MM * HNN * 4);
  float* gv_pre  = (float*)alloc((size_t)MM * KVD_ * 4);
  float* gk_pre  = (float*)alloc((size_t)MM * KVD_ * 4);
  float* gbuf    = (float*)alloc((size_t)MM * HNN * 4);
  float* rB      = (float*)alloc((size_t)MM * HNN * 4);
  float* decayB  = (float*)alloc((size_t)MM * HNN * 4);
  float* kB      = (float*)alloc((size_t)MM * HNN * 4);
  float* vB      = (float*)alloc((size_t)MM * HNN * 4);
  float* aaB     = (float*)alloc((size_t)MM * HNN * 4);
  float* bbB     = (float*)alloc((size_t)MM * HNN * 4);
  float* bonusB  = (float*)alloc((size_t)MM * HNN * 4);
  float* obuf    = (float*)alloc((size_t)MM * HNN * 4);

  auto blocks = [](int n) { return (n + 255) / 256; };

  // weight casts (NT weights straight, NN weights transposed to NT)
  cast_bf16_kernel<<<blocks(RKVN * HNN), 256, 0, stream>>>(RKV_w, wrkv_bf, RKVN * HNN);
  cast_bf16_kernel<<<blocks(HNN * HNN), 256, 0, stream>>>(O_w, owbf, HNN * HNN);
  cast_bf16_tr_kernel<<<blocks(HNN * 320), 256, 0, stream>>>(wavgk1, wavgT, HNN, 320);
  cast_bf16_tr_kernel<<<blocks(64 * HNN), 256, 0, stream>>>(w2, w2T, 64, HNN);
  cast_bf16_tr_kernel<<<blocks(64 * HNN), 256, 0, stream>>>(a2, a2T, 64, HNN);
  cast_bf16_tr_kernel<<<blocks(32 * KVD_), 256, 0, stream>>>(v2, v2T, 32, KVD_);
  cast_bf16_tr_kernel<<<blocks(128 * HNN), 256, 0, stream>>>(g2, g2T, 128, HNN);
  cast_bf16_tr_kernel<<<blocks(32 * KVD_), 256, 0, stream>>>(k2, k2T, 32, KVD_);
  concat_bias_kernel<<<blocks(RKVN), 256, 0, stream>>>(R_bias, K_bias, V_bias, biascat);

  // x = RMS(x_in, ln1) -> bf16
  rms_x_kernel<<<MM, 256, 0, stream>>>(x_in, ln1, xbf);

  // big GEMMs (WMMA)
  launch_gemm(xbf, HNN, wrkv_bf, HNN, biascat, nullptr, rkv, RKVN, HNN, RKVN, stream);
  launch_gemm(xbf, HNN, wavgT, HNN, nullptr, nullptr, xwavgk, 320, HNN, 320, stream);

  // LoRA inputs with pre-activations, then LoRA GEMMs (WMMA)
  act_split_kernel<<<blocks(MM * 320), 256, 0, stream>>>(xwavgk, xw_bf, xa_bf, xv_bf, xg_bf, xk_bf);
  launch_gemm(xw_bf, 64, w2T, 64, w0, nullptr, w_pre, HNN, 64, HNN, stream);
  launch_gemm(xa_bf, 64, a2T, 64, a0, nullptr, a_pre, HNN, 64, HNN, stream);
  launch_gemm(xv_bf, 32, v2T, 32, v0, nullptr, gv_pre, KVD_, 32, KVD_, stream);
  launch_gemm(xk_bf, 32, k2T, 32, k0, nullptr, gk_pre, KVD_, 32, KVD_, stream);
  launch_gemm(xg_bf, 128, g2T, 128, nullptr, nullptr, gbuf, HNN, 128, HNN, stream);

  // per-token prep
  prep_kernel<<<MM, 256, 0, stream>>>(rkv, w_pre, a_pre, gv_pre, gk_pre, v_first, k_first,
                                      ccos, csin, R_bias, K_bias, V_bias, ln_r, ln_k, r_k,
                                      rB, decayB, kB, vB, aaB, bbB, bonusB);

  // sequential scan over T (64 independent (b,h) blocks, async-LDS staging)
  scan_kernel<<<BB * HH, 256, 0, stream>>>(rB, decayB, kB, vB, aaB, bbB, state, obuf);

  // y = (o/sqrt(N) + bonus) * g -> bf16 ; out = x_in + y @ O_w^T + O_bias (WMMA, fused)
  mixg_kernel<<<blocks(MM * HNN), 256, 0, stream>>>(obuf, bonusB, gbuf, ybf, MM * HNN);
  launch_gemm(ybf, HNN, owbf, HNN, O_bias, x_in, out, HNN, HNN, HNN, stream);
}
